// BidirectionalCrossLevelAttention_77386720740038
// MI455X (gfx1250) — matrix operand order repaired and checked by reading, not compiled, not run on hardware
//
#include <hip/hip_runtime.h>
#include <hip/hip_bf16.h>

// ---------------------------------------------------------------------------
// MI455X (gfx1250): WMMA bf16 GEMMs + flash attention with async global->LDS
// ---------------------------------------------------------------------------

typedef __attribute__((ext_vector_type(16))) __bf16 v16bf;
typedef __attribute__((ext_vector_type(8)))  float  v8f;

union FragB { v16bf v; uint4 u[2]; };

__device__ __forceinline__ unsigned short f2bf(float f) {
  union { float f; unsigned u; } c; c.f = f;
  unsigned r = c.u + 0x7FFFu + ((c.u >> 16) & 1u);   // round-to-nearest-even
  return (unsigned short)(r >> 16);
}

struct EpiParams {
  const float* bias;   // optional
  const float* cin;    // EPI 2: pre-accumulated other-half of gate GEMM
  const float* a2;     // EPI 2: aoLN ; EPI 3: ao2row (256x512)
  const float* x;      // EPI 2: h_tissue ; EPI 3: h_cell
  const float* gp;     // EPI 3: gate precomp (256x512)
  const int*   idx;    // EPI 3: k_idx per row
  const int*   has;    // EPI 2: has_cells per row
};

// Generic C[M,N] = A[M,K] * B[K,N] (+bias) with epilogues.
// EPI 0: f32 store (+bias)      EPI 1: bf16 store (+bias)
// EPI 2: gate_bu: g=sig(acc+cin); out = has? g*a2+(1-g)*x : x
// EPI 3: gate_td: g=sig(acc+gp[idx]); out = g*a2[idx]+(1-g)*x
template <int EPI>
__global__ __launch_bounds__(128)
void wmma_gemm(const float* __restrict__ A, const float* __restrict__ B,
               void* __restrict__ Cout, int M, int N, int Kd, EpiParams ep)
{
  __shared__ __align__(16) unsigned short a_lds[64][40];  // row-major bf16
  __shared__ __align__(16) unsigned short b_lds[64][40];  // B^T: [n][k]
  const int t    = threadIdx.x;
  const int wave = t >> 5, lane = t & 31, hi = lane >> 4, lm = lane & 15;
  const int rowBase = blockIdx.x * 64;
  const int colBase = blockIdx.y * 64;

  v8f acc[4] = {};

  for (int k0 = 0; k0 < Kd; k0 += 32) {
    // stage A tile 64x32 (f32 -> bf16)
#pragma unroll
    for (int i = 0; i < 4; ++i) {
      int e = t + i * 128;                 // float4 units, 0..511
      int r = e >> 3, c4 = e & 7;
      int gr = rowBase + r; if (gr >= M) gr = M - 1;
      float4 v = *(const float4*)(A + (size_t)gr * Kd + k0 + c4 * 4);
      unsigned short* dst = &a_lds[r][c4 * 4];
      dst[0] = f2bf(v.x); dst[1] = f2bf(v.y); dst[2] = f2bf(v.z); dst[3] = f2bf(v.w);
    }
    // stage B tile 32x64 transposed into b_lds[n][k]
#pragma unroll
    for (int i = 0; i < 4; ++i) {
      int e = t + i * 128;
      int kr = e >> 4, c4 = e & 15;
      float4 v = *(const float4*)(B + (size_t)(k0 + kr) * N + colBase + c4 * 4);
      b_lds[c4 * 4 + 0][kr] = f2bf(v.x); b_lds[c4 * 4 + 1][kr] = f2bf(v.y);
      b_lds[c4 * 4 + 2][kr] = f2bf(v.z); b_lds[c4 * 4 + 3][kr] = f2bf(v.w);
    }
    if (k0 + 32 < Kd) {  // hint next A tile -> global_prefetch
      int gr = rowBase + (t & 63); if (gr >= M) gr = M - 1;
      __builtin_prefetch(A + (size_t)gr * Kd + k0 + 32, 0, 1);
    }
    __syncthreads();

    // A fragment: rows wave*16..+15, contiguous k = 16*hi + s (same order as B)
    FragB fa;
    {
      const unsigned short* ar = &a_lds[wave * 16 + lm][0];
      fa.u[0] = *(const uint4*)(ar + hi * 16);
      fa.u[1] = *(const uint4*)(ar + hi * 16 + 8);
    }
#pragma unroll
    for (int nt = 0; nt < 4; ++nt) {
      FragB fb;
      const unsigned short* br = &b_lds[nt * 16 + lm][0];
      fb.u[0] = *(const uint4*)(br + hi * 16);
      fb.u[1] = *(const uint4*)(br + hi * 16 + 8);
      acc[nt] = __builtin_amdgcn_wmma_f32_16x16x32_bf16(
          false, fa.v, false, fb.v, (short)0, acc[nt], false, false);
    }
    __syncthreads();
  }

  // epilogue (C layout: lane<16 -> M=r, lane>=16 -> M=8+r; N = lane&15)
#pragma unroll
  for (int nt = 0; nt < 4; ++nt) {
    int col = colBase + nt * 16 + lm;
#pragma unroll
    for (int r = 0; r < 8; ++r) {
      int row = rowBase + wave * 16 + (hi ? 8 + r : r);
      if (row >= M) continue;
      float v = acc[nt][r];
      size_t off = (size_t)row * N + col;
      if (EPI == 0) {
        if (ep.bias) v += ep.bias[col];
        ((float*)Cout)[off] = v;
      } else if (EPI == 1) {
        if (ep.bias) v += ep.bias[col];
        ((unsigned short*)Cout)[off] = f2bf(v);
      } else if (EPI == 2) {
        v += ep.cin[off];
        float g  = 1.f / (1.f + __expf(-v));
        float xx = ep.x[off];
        float upd = g * ep.a2[off] + (1.f - g) * xx;
        ((float*)Cout)[off] = ep.has[row] ? upd : xx;
      } else {
        int kix = ep.idx[row];
        size_t ko = (size_t)kix * N + col;
        v += ep.gp[ko];
        float g = 1.f / (1.f + __expf(-v));
        ((float*)Cout)[off] = g * ep.a2[ko] + (1.f - g) * ep.x[off];
      }
    }
  }
}

// ---------------------------------------------------------------------------
// Flash attention (bottom-up): per block = (slice, h*2+qhalf); 128 queries,
// dk=128, masked online softmax over key tiles of 32 using WMMA for QK^T & PV.
// K tile is copied global->LDS with GLOBAL_LOAD_ASYNC_TO_LDS_B128 (ASYNCcnt),
// overlapping with the VALU-heavy transposed V staging.
// ---------------------------------------------------------------------------
#define NSLICES 8

__global__ __launch_bounds__(256)
void flash_attn(const unsigned short* __restrict__ kh, const unsigned short* __restrict__ vh,
                const float* __restrict__ qh, const float* __restrict__ Sin,
                float* __restrict__ partO, float* __restrict__ partM, float* __restrict__ partL,
                int N, int tilesPerSlice, int totalTiles)
{
  __shared__ __align__(16) unsigned short q_lds[128][136];
  __shared__ __align__(16) unsigned short kt_lds[32][136];
  __shared__ __align__(16) unsigned short vt_lds[128][40];   // V^T [dk][key]
  __shared__ __align__(16) unsigned short p_lds[8][16][40];  // per-wave P

  const int t = threadIdx.x;
  const int wave = t >> 5, lane = t & 31, hi = lane >> 4, lm = lane & 15;
  const int slice = blockIdx.x;
  const int hq = blockIdx.y, h = hq >> 1, half = hq & 1;
  const int qBase = half * 128;
  const float scale = 0.08838834764831845f;  // 1/sqrt(128)

  // stage Q (f32 -> bf16)
#pragma unroll
  for (int i = 0; i < 16; ++i) {
    int e = t + i * 256;                      // float4 units, 0..4095
    int r = e >> 5, c4 = e & 31;
    float4 v = *(const float4*)(qh + (size_t)(qBase + r) * 512 + h * 128 + c4 * 4);
    unsigned short* dst = &q_lds[r][c4 * 4];
    dst[0] = f2bf(v.x); dst[1] = f2bf(v.y); dst[2] = f2bf(v.z); dst[3] = f2bf(v.w);
  }
  __syncthreads();

  FragB aQ[4];
#pragma unroll
  for (int c = 0; c < 4; ++c) {
    const unsigned short* qr = &q_lds[wave * 16 + lm][c * 32];
    aQ[c].u[0] = *(const uint4*)(qr + hi * 16);
    aQ[c].u[1] = *(const uint4*)(qr + hi * 16 + 8);
  }

  v8f o[8] = {};
  float m_run[8], l_run[8];
#pragma unroll
  for (int r = 0; r < 8; ++r) { m_run[r] = -1e9f; l_run[r] = 0.f; }

  int t0 = slice * tilesPerSlice;
  int t1 = t0 + tilesPerSlice; if (t1 > totalTiles) t1 = totalTiles;

  for (int tt = t0; tt < t1; ++tt) {
    const int keyBase = tt * 32;
    const bool fullTile = (keyBase + 32 <= N);   // uniform across block

    // --- K tile (32x128 bf16, verbatim copy): async DMA into LDS ---
    if (fullTile) {
#pragma unroll
      for (int i = 0; i < 2; ++i) {
        int e = t + i * 256;                    // b128 chunk id
        int key = e >> 4, c8 = e & 15;
        const unsigned short* gp = kh + (size_t)(keyBase + key) * 512 + h * 128 + c8 * 8;
        unsigned ldsa = (unsigned)(uintptr_t)(&kt_lds[key][c8 * 8]);
        asm volatile("global_load_async_to_lds_b128 %0, %1, off"
                     :: "v"(ldsa), "v"(gp) : "memory");
      }
    } else {
#pragma unroll
      for (int i = 0; i < 2; ++i) {
        int e = t + i * 256;
        int key = e >> 4, c8 = e & 15;
        int n = keyBase + key;
        uint4 v = make_uint4(0, 0, 0, 0);
        if (n < N) v = *(const uint4*)(kh + (size_t)n * 512 + h * 128 + c8 * 8);
        *(uint4*)&kt_lds[key][c8 * 8] = v;
      }
    }
    // --- V tile transposed -> vt[dk][key] (overlaps with async K copy) ---
#pragma unroll
    for (int i = 0; i < 16; ++i) {
      int e = t + i * 256;
      int key = e >> 7, dk = e & 127;
      int n = keyBase + key;
      unsigned short v = 0;
      if (n < N) v = vh[(size_t)n * 512 + h * 128 + dk];
      vt_lds[dk][key] = v;
    }
    if (fullTile)
      asm volatile("s_wait_asynccnt 0x0" ::: "memory");  // K copy visible pre-barrier
    __syncthreads();

    // S = Q K^T (16q x 32keys per wave)
    v8f s0 = {}, s1 = {};
#pragma unroll
    for (int c = 0; c < 4; ++c) {
      FragB fb;
      const unsigned short* k0p = &kt_lds[lm][c * 32];
      fb.u[0] = *(const uint4*)(k0p + hi * 16);
      fb.u[1] = *(const uint4*)(k0p + hi * 16 + 8);
      s0 = __builtin_amdgcn_wmma_f32_16x16x32_bf16(false, aQ[c].v, false, fb.v, (short)0, s0, false, false);
      const unsigned short* k1p = &kt_lds[16 + lm][c * 32];
      fb.u[0] = *(const uint4*)(k1p + hi * 16);
      fb.u[1] = *(const uint4*)(k1p + hi * 16 + 8);
      s1 = __builtin_amdgcn_wmma_f32_16x16x32_bf16(false, aQ[c].v, false, fb.v, (short)0, s1, false, false);
    }

    // mask + online softmax (row stats across 16-lane groups)
    float corr[8];
#pragma unroll
    for (int r = 0; r < 8; ++r) {
      int rowLoc = hi ? 8 + r : r;
      int qg = qBase + wave * 16 + rowLoc;
      int n0 = keyBase + lm, n1 = keyBase + 16 + lm;
      float v0 = s0[r] * scale, v1 = s1[r] * scale;
      bool ok0 = (n0 < N) && (Sin[(size_t)n0 * 256 + qg] > 0.1f);
      bool ok1 = (n1 < N) && (Sin[(size_t)n1 * 256 + qg] > 0.1f);
      v0 = ok0 ? v0 : -1e9f;
      v1 = ok1 ? v1 : -1e9f;
      float mx = fmaxf(v0, v1);
      for (int d = 8; d >= 1; d >>= 1) mx = fmaxf(mx, __shfl_xor(mx, d, 16));
      float mnew = fmaxf(m_run[r], mx);
      float c0 = __expf(m_run[r] - mnew);
      float p0 = __expf(v0 - mnew), p1 = __expf(v1 - mnew);
      float ps = p0 + p1;
      for (int d = 8; d >= 1; d >>= 1) ps += __shfl_xor(ps, d, 16);
      l_run[r] = l_run[r] * c0 + ps;
      m_run[r] = mnew;
      corr[r] = c0;
      s0[r] = p0; s1[r] = p1;
    }
#pragma unroll
    for (int nt = 0; nt < 8; ++nt)
#pragma unroll
      for (int r = 0; r < 8; ++r) o[nt][r] *= corr[r];

    // P -> LDS (C-layout to A-layout bounce), then O += P V
#pragma unroll
    for (int r = 0; r < 8; ++r) {
      int rowLoc = hi ? 8 + r : r;
      p_lds[wave][rowLoc][lm]      = f2bf(s0[r]);
      p_lds[wave][rowLoc][16 + lm] = f2bf(s1[r]);
    }
    FragB ap;
    {
      const unsigned short* pr = &p_lds[wave][lm][0];
      ap.u[0] = *(const uint4*)(pr + hi * 16);
      ap.u[1] = *(const uint4*)(pr + hi * 16 + 8);
    }
#pragma unroll
    for (int nt = 0; nt < 8; ++nt) {
      FragB fb;
      const unsigned short* vr = &vt_lds[nt * 16 + lm][0];
      fb.u[0] = *(const uint4*)(vr + hi * 16);
      fb.u[1] = *(const uint4*)(vr + hi * 16 + 8);
      o[nt] = __builtin_amdgcn_wmma_f32_16x16x32_bf16(false, ap.v, false, fb.v, (short)0, o[nt], false, false);
    }
    __syncthreads();
  }

  // store partial (O unnormalized, m, l)
  const int pb = hq * gridDim.x + slice;
#pragma unroll
  for (int nt = 0; nt < 8; ++nt) {
    int dk = nt * 16 + lm;
#pragma unroll
    for (int r = 0; r < 8; ++r) {
      int qloc = wave * 16 + (hi ? 8 + r : r);
      partO[((size_t)pb * 128 + qloc) * 128 + dk] = o[nt][r];
    }
  }
  if (lm == 0) {
#pragma unroll
    for (int r = 0; r < 8; ++r) {
      int qloc = wave * 16 + (hi ? 8 + r : r);
      partM[(size_t)pb * 128 + qloc] = m_run[r];
      partL[(size_t)pb * 128 + qloc] = l_run[r];
    }
  }
}

__global__ __launch_bounds__(128)
void flash_merge(const float* __restrict__ partO, const float* __restrict__ partM,
                 const float* __restrict__ partL, float* __restrict__ ao, int nslices)
{
  int h = blockIdx.x >> 8, q = blockIdx.x & 255;
  int half = q >> 7, qloc = q & 127;
  int hq = h * 2 + half;
  float mstar = -3e38f;
  for (int s = 0; s < nslices; ++s)
    mstar = fmaxf(mstar, partM[(size_t)(hq * nslices + s) * 128 + qloc]);
  float lstar = 0.f;
  for (int s = 0; s < nslices; ++s) {
    size_t pb = (size_t)(hq * nslices + s) * 128 + qloc;
    lstar += partL[pb] * __expf(partM[pb] - mstar);
  }
  int d = threadIdx.x;
  float oacc = 0.f;
  for (int s = 0; s < nslices; ++s) {
    size_t pb = (size_t)(hq * nslices + s) * 128 + qloc;
    oacc += partO[pb * 128 + d] * __expf(partM[pb] - mstar);
  }
  float inv = lstar > 0.f ? 1.f / lstar : 0.f;
  ao[(size_t)q * 512 + h * 128 + d] = oacc * inv;
}

// ---------------------------------------------------------------------------
// Small helper kernels
// ---------------------------------------------------------------------------
__global__ void bias_combine(const float* __restrict__ bprev, const float* __restrict__ W,
                             const float* __restrict__ badd, float* __restrict__ out, int Kd, int Nd)
{
  int n = blockIdx.x * blockDim.x + threadIdx.x;
  if (n >= Nd) return;
  float acc = badd ? badd[n] : 0.f;
  if (bprev)
    for (int k = 0; k < Kd; ++k) acc += bprev[k] * W[(size_t)k * Nd + n];
  out[n] = acc;
}

__global__ __launch_bounds__(128)
void layernorm_rows(const float* __restrict__ X, const float* __restrict__ g,
                    const float* __restrict__ b, float* __restrict__ Y)
{
  __shared__ float s1[128], s2[128];
  int row = blockIdx.x, t = threadIdx.x;
  float a0 = 0.f, a1 = 0.f, vals[4];
#pragma unroll
  for (int i = 0; i < 4; ++i) {
    float v = X[(size_t)row * 512 + t + i * 128];
    vals[i] = v; a0 += v; a1 += v * v;
  }
  s1[t] = a0; s2[t] = a1; __syncthreads();
  for (int off = 64; off > 0; off >>= 1) {
    if (t < off) { s1[t] += s1[t + off]; s2[t] += s2[t + off]; }
    __syncthreads();
  }
  float mean = s1[0] * (1.f / 512.f);
  float var  = s2[0] * (1.f / 512.f) - mean * mean;
  float inv  = rsqrtf(var + 1e-5f);
#pragma unroll
  for (int i = 0; i < 4; ++i) {
    int c = t + i * 128;
    Y[(size_t)row * 512 + c] = (vals[i] - mean) * inv * g[c] + b[c];
  }
}

__global__ void argmax_rows(const float* __restrict__ S, int* __restrict__ idx, int Nrows)
{
  int n = blockIdx.x * blockDim.x + threadIdx.x;
  if (n >= Nrows) return;
  const float* r = S + (size_t)n * 256;
  float best = r[0]; int bi = 0;
  for (int k = 1; k < 256; ++k) { float v = r[k]; if (v > best) { best = v; bi = k; } }
  idx[n] = bi;
}

__global__ __launch_bounds__(256)
void has_cells_kernel(const float* __restrict__ S, int* __restrict__ has, int N)
{
  __shared__ unsigned mword;
  if (threadIdx.x == 0) mword = 0;
  __syncthreads();
  int k0 = blockIdx.x * 32;
  unsigned m = 0;
  for (int n = threadIdx.x; n < N; n += 256) {
    const float4* p = (const float4*)(S + (size_t)n * 256 + k0);
#pragma unroll
    for (int j = 0; j < 8; ++j) {
      float4 v = p[j];
      if (v.x > 0.1f) m |= 1u << (j * 4 + 0);
      if (v.y > 0.1f) m |= 1u << (j * 4 + 1);
      if (v.z > 0.1f) m |= 1u << (j * 4 + 2);
      if (v.w > 0.1f) m |= 1u << (j * 4 + 3);
    }
  }
  atomicOr(&mword, m);
  __syncthreads();
  if (threadIdx.x < 32) has[k0 + threadIdx.x] = (mword >> threadIdx.x) & 1;
}

// ---------------------------------------------------------------------------
// Orchestration
// ---------------------------------------------------------------------------
extern "C" void kernel_launch(void* const* d_in, const int* in_sizes, int n_in,
                              void* d_out, int out_size, void* d_ws, size_t ws_size,
                              hipStream_t stream)
{
  (void)n_in; (void)out_size; (void)ws_size;
  const int N = in_sizes[0] / 512;           // 50000 cells
  const float* h_cell   = (const float*)d_in[0];
  const float* h_tissue = (const float*)d_in[1];
  const float* Smat     = (const float*)d_in[2];
  const float* P[36];
  for (int i = 0; i < 36; ++i) P[i] = (const float*)d_in[3 + i];
  // param order (setup_inputs insertion order):
  const float *w_bu_q=P[0], *b_bu_q=P[1], *w_bu_k=P[2], *b_bu_k=P[3], *w_bu_v=P[4], *b_bu_v=P[5];
  const float *w_td_v=P[10], *b_td_v=P[11];
  const float *bu_wq=P[12], *bu_bq=P[13], *bu_wk=P[14], *bu_bk=P[15], *bu_wv=P[16], *bu_bv=P[17];
  const float *bu_wo=P[18], *bu_bo=P[19], *bu_ln_g=P[20], *bu_ln_b=P[21];
  const float *td_wv=P[26], *td_bv=P[27], *td_wo=P[28], *td_bo=P[29], *td_ln_g=P[30], *td_ln_b=P[31];
  const float *gate_bu_w=P[32], *gate_bu_b=P[33], *gate_td_w=P[34], *gate_td_b=P[35];

  float* wsf = (float*)d_ws;
  size_t o = 0;
  auto alloc = [&](size_t n) { size_t r = o; o += (n + 63) & ~(size_t)63; return r; };
  size_t oWq = alloc(512 * 512), oWk = alloc(512 * 512), oWv = alloc(512 * 512);
  size_t oWt1 = alloc(512 * 512), oWtd = alloc(512 * 512);
  size_t obq = alloc(512), obk = alloc(512), obv = alloc(512), obt1 = alloc(512), obtd = alloc(512);
  size_t oqh = alloc(256 * 512);
  size_t okh = alloc((size_t)N * 256);       // bf16: N*512 ushorts
  size_t ovh = alloc((size_t)N * 256);
  size_t oao  = alloc(256 * 512), oaoP = alloc(256 * 512), oaoLN = alloc(256 * 512);
  size_t otmp = alloc(256 * 512), oa2p = alloc(256 * 512), oa2 = alloc(256 * 512), ogp = alloc(256 * 512);
  size_t opO = alloc((size_t)8 * NSLICES * 128 * 128);
  size_t opM = alloc(8 * NSLICES * 128), opL = alloc(8 * NSLICES * 128);
  size_t okidx = alloc(N), ohas = alloc(256);

  float* out_cell   = (float*)d_out;
  float* out_tissue = (float*)d_out + (size_t)N * 512;

  EpiParams e0 = {}; // plain
  dim3 blk(128);
  dim3 gW(8, 8), g256(4, 8), gN((N + 63) / 64, 8);

  // 1) combine chained projection weights (on-device 512^3 WMMA GEMMs)
  { EpiParams e = e0; wmma_gemm<0><<<gW, blk, 0, stream>>>(w_bu_q, bu_wq, wsf + oWq, 512, 512, 512, e); }
  { EpiParams e = e0; wmma_gemm<0><<<gW, blk, 0, stream>>>(w_bu_k, bu_wk, wsf + oWk, 512, 512, 512, e); }
  { EpiParams e = e0; wmma_gemm<0><<<gW, blk, 0, stream>>>(w_bu_v, bu_wv, wsf + oWv, 512, 512, 512, e); }
  { EpiParams e = e0; wmma_gemm<0><<<gW, blk, 0, stream>>>(w_td_v, td_wv, wsf + oWt1, 512, 512, 512, e); }
  { EpiParams e = e0; wmma_gemm<0><<<gW, blk, 0, stream>>>(wsf + oWt1, td_wo, wsf + oWtd, 512, 512, 512, e); }
  bias_combine<<<2, 256, 0, stream>>>(b_bu_q, bu_wq, bu_bq, wsf + obq, 512, 512);
  bias_combine<<<2, 256, 0, stream>>>(b_bu_k, bu_wk, bu_bk, wsf + obk, 512, 512);
  bias_combine<<<2, 256, 0, stream>>>(b_bu_v, bu_wv, bu_bv, wsf + obv, 512, 512);
  bias_combine<<<2, 256, 0, stream>>>(b_td_v, td_wv, td_bv, wsf + obt1, 512, 512);
  bias_combine<<<2, 256, 0, stream>>>(wsf + obt1, td_wo, td_bo, wsf + obtd, 512, 512);

  // 2) projections: qh (256x512 f32), kh/vh (Nx512 bf16)
  { EpiParams e = e0; e.bias = wsf + obq; wmma_gemm<0><<<g256, blk, 0, stream>>>(h_tissue, wsf + oWq, wsf + oqh, 256, 512, 512, e); }
  { EpiParams e = e0; e.bias = wsf + obk; wmma_gemm<1><<<gN, blk, 0, stream>>>(h_cell, wsf + oWk, wsf + okh, N, 512, 512, e); }
  { EpiParams e = e0; e.bias = wsf + obv; wmma_gemm<1><<<gN, blk, 0, stream>>>(h_cell, wsf + oWv, wsf + ovh, N, 512, 512, e); }

  // 3) mask occupancy + flash attention + merge -> ao (256x512)
  has_cells_kernel<<<8, 256, 0, stream>>>(Smat, (int*)(wsf + ohas), N);
  int totalTiles = (N + 31) / 32;
  int tilesPerSlice = (totalTiles + NSLICES - 1) / NSLICES;
  flash_attn<<<dim3(NSLICES, 8), 256, 0, stream>>>(
      (const unsigned short*)(wsf + okh), (const unsigned short*)(wsf + ovh),
      wsf + oqh, Smat, wsf + opO, wsf + opM, wsf + opL, N, tilesPerSlice, totalTiles);
  flash_merge<<<1024, 128, 0, stream>>>(wsf + opO, wsf + opM, wsf + opL, wsf + oao, NSLICES);

  // 4) bottom-up output proj + LN + gated tissue update -> h_tissue_new
  { EpiParams e = e0; e.bias = bu_bo; wmma_gemm<0><<<g256, blk, 0, stream>>>(wsf + oao, bu_wo, wsf + oaoP, 256, 512, 512, e); }
  layernorm_rows<<<256, 128, 0, stream>>>(wsf + oaoP, bu_ln_g, bu_ln_b, wsf + oaoLN);
  { EpiParams e = e0; e.bias = gate_bu_b; wmma_gemm<0><<<g256, blk, 0, stream>>>(h_tissue, gate_bu_w, wsf + otmp, 256, 512, 512, e); }
  { EpiParams e = e0; e.cin = wsf + otmp; e.a2 = wsf + oaoLN; e.x = h_tissue; e.has = (const int*)(wsf + ohas);
    wmma_gemm<2><<<g256, blk, 0, stream>>>(wsf + oaoLN, gate_bu_w + (size_t)512 * 512, out_tissue, 256, 512, 512, e); }

  // 5) top-down: per-region (256 rows!) collapsed chain, then one N-GEMM w/ gather epilogue
  { EpiParams e = e0; e.bias = wsf + obtd; wmma_gemm<0><<<g256, blk, 0, stream>>>(out_tissue, wsf + oWtd, wsf + oa2p, 256, 512, 512, e); }
  layernorm_rows<<<256, 128, 0, stream>>>(wsf + oa2p, td_ln_g, td_ln_b, wsf + oa2);
  { EpiParams e = e0; e.bias = gate_td_b; wmma_gemm<0><<<g256, blk, 0, stream>>>(wsf + oa2, gate_td_w + (size_t)512 * 512, wsf + ogp, 256, 512, 512, e); }
  argmax_rows<<<(N + 255) / 256, 256, 0, stream>>>(Smat, (int*)(wsf + okidx), N);
  { EpiParams e = e0; e.gp = wsf + ogp; e.a2 = wsf + oa2; e.x = h_cell; e.idx = (const int*)(wsf + okidx);
    wmma_gemm<3><<<gN, blk, 0, stream>>>(h_cell, gate_td_w, out_cell, N, 512, 512, e); }
}
